// MultiHeadSelfAttention_66099546685456
// MI455X (gfx1250) — matrix operand order repaired and checked by reading, not compiled, not run on hardware
//
#include <hip/hip_runtime.h>
#include <hip/hip_bf16.h>

typedef __attribute__((ext_vector_type(16))) _Float16 v16h;
typedef __attribute__((ext_vector_type(8)))  float    v8f;

#define D_MODEL   1024
#define NUM_HEADS 16
#define HEAD_DIM  64
#define SEQ       2048
#define BATCH     4
#define ROWS      (BATCH * SEQ)    // 8192
#define N_QKV     (3 * D_MODEL)    // 3072

// LDS row pitches (halves), padded to reduce DS bank conflicts
#define KL_PITCH  72               // 32 rows (keys) x 64 d
#define VL_PITCH  40               // 64 rows (d)   x 32 t

// ---------------------------------------------------------------------------
// WMMA fragment loaders (contiguous-K order, same for A and B, wave32)
// ---------------------------------------------------------------------------

// A fragment: 16(M) x 32(K) f16. Lane L: row = L%16; element e maps to
// K = 16*(L/16) + e.  -> one contiguous 16-half chunk per lane.
__device__ __forceinline__ v16h load_a_f16(const _Float16* __restrict__ A, int ld,
                                           int m0, int k0, int lane) {
  const int m  = m0 + (lane & 15);
  const int kb = k0 + 16 * (lane >> 4);
  const _Float16* p = A + (size_t)m * ld + kb;
  v16h a;
#pragma unroll
  for (int i = 0; i < 8; ++i) { a[i] = p[i]; a[8 + i] = p[8 + i]; }
  return a;
}

// Same, but source is fp32 (convert on the fly).
__device__ __forceinline__ v16h load_a_f32(const float* __restrict__ A, int ld,
                                           int m0, int k0, int lane) {
  const int m  = m0 + (lane & 15);
  const int kb = k0 + 16 * (lane >> 4);
  const float* p = A + (size_t)m * ld + kb;
  v16h a;
#pragma unroll
  for (int i = 0; i < 8; ++i) { a[i] = (_Float16)p[i]; a[8 + i] = (_Float16)p[8 + i]; }
  return a;
}

// B fragment: 32(K) x 16(N) f16, supplied as its transpose Bt[N][K] row-major
// (leading dim ld). Lane L: col = L%16; element e -> K = k0 + 16*(L/16) + e.
// -> one contiguous 16-half (32B) chunk per lane.
__device__ __forceinline__ v16h load_bT_f16(const _Float16* __restrict__ Bt, int ld,
                                            int k0, int n0, int lane) {
  const int n  = n0 + (lane & 15);
  const int kb = k0 + 16 * (lane >> 4);
  const _Float16* p = Bt + (size_t)n * ld + kb;
  v16h b;
#pragma unroll
  for (int i = 0; i < 16; ++i) b[i] = p[i];
  return b;
}

__device__ __forceinline__ v8f wmma_f16(v16h a, v16h b, v8f c) {
  return __builtin_amdgcn_wmma_f32_16x16x32_f16(false, a, false, b, (short)0, c,
                                                false, false);
}

// ---------------------------------------------------------------------------
// Stage 1: fp32 -> f16 transpose of weights: dst[C][R] = src[R][C]
// ---------------------------------------------------------------------------
__global__ void __launch_bounds__(256)
transpose_f32_to_f16(const float* __restrict__ src, _Float16* __restrict__ dst,
                     int R, int C) {
  int idx = blockIdx.x * blockDim.x + threadIdx.x;
  if (idx >= R * C) return;
  int c = idx / R;
  int r = idx - c * R;
  dst[idx] = (_Float16)src[(size_t)r * C + c];  // idx == c*R + r (coalesced store)
}

// ---------------------------------------------------------------------------
// Stage 2: QKV projection. One wave -> 16x64 output tile, K = 1024.
// Writes Q[b,h,t,d], K[b,h,t,d] and transposed Vt[b,h,d,t] in f16.
// ---------------------------------------------------------------------------
__global__ void __launch_bounds__(256)
gemm_qkv_kernel(const float* __restrict__ x, const _Float16* __restrict__ wT,
                const float* __restrict__ bias,
                _Float16* __restrict__ Qh, _Float16* __restrict__ Kh,
                _Float16* __restrict__ Vth) {
  const int lane = threadIdx.x & 31;
  const int wave = threadIdx.x >> 5;
  const int tile = blockIdx.x * 8 + wave;     // 24576 wave-tiles
  const int NT = N_QKV / 64;                  // 48
  const int mt = tile / NT, nt = tile - mt * NT;
  const int m0 = mt * 16, n0 = nt * 64;

  v8f zero = {};
  v8f acc[4] = {zero, zero, zero, zero};
  for (int k0 = 0; k0 < D_MODEL; k0 += 32) {
    v16h a = load_a_f32(x, D_MODEL, m0, k0, lane);
#pragma unroll
    for (int g = 0; g < 4; ++g) {
      v16h b = load_bT_f16(wT, D_MODEL, k0, n0 + g * 16, lane);
      acc[g] = wmma_f16(a, b, acc[g]);
    }
  }

  const int half = lane >> 4;
  const int nl   = lane & 15;
  const int sect = n0 / D_MODEL;              // 0:Q 1:K 2:V (tile never straddles)
  const int h    = (n0 % D_MODEL) / HEAD_DIM; // head fixed per tile (64-wide tile)
#pragma unroll
  for (int g = 0; g < 4; ++g) {
    const int d = g * 16 + nl;
    const float bv = bias[n0 + g * 16 + nl];
#pragma unroll
    for (int r = 0; r < 8; ++r) {
      const int m  = m0 + r + 8 * half;
      const int t  = m & (SEQ - 1);
      const int bb = m >> 11;                 // m / SEQ
      const _Float16 val = (_Float16)(acc[g][r] + bv);
      const size_t head = (size_t)(bb * NUM_HEADS + h);
      if (sect == 0)      Qh [(head * SEQ + t) * HEAD_DIM + d] = val;
      else if (sect == 1) Kh [(head * SEQ + t) * HEAD_DIM + d] = val;
      else                Vth[(head * HEAD_DIM + d) * SEQ + t] = val;
    }
  }
}

// ---------------------------------------------------------------------------
// Stage 3: block-cooperative flash attention.
// One block (4 waves) owns 64 query rows of one (b,h); each 32-key chunk of
// K and V^T is staged into LDS ONCE and shared by all 4 waves (4x less L2
// traffic than per-wave streaming: ~4 GB -> ~1 GB).
// ---------------------------------------------------------------------------
__global__ void __launch_bounds__(128)
attn_kernel(const _Float16* __restrict__ Qh, const _Float16* __restrict__ Kh,
            const _Float16* __restrict__ Vth, _Float16* __restrict__ atth) {
  __shared__ _Float16 Kl[32 * KL_PITCH];   // keys x d   (4.5 KB)
  __shared__ _Float16 Vl[64 * VL_PITCH];   // d    x t   (5.0 KB)
  __shared__ _Float16 Pl[4][16 * 32];      // per-wave P (4.0 KB)

  const int tid  = threadIdx.x;
  const int lane = tid & 31;
  const int wave = tid >> 5;
  const int bh = blockIdx.x >> 5;             // [0,64)
  const int qb = blockIdx.x & 31;             // 32 query blocks of 64 rows
  const int b  = bh >> 4, h = bh & 15;

  const _Float16* Q  = Qh  + (size_t)bh * SEQ * HEAD_DIM;
  const _Float16* K  = Kh  + (size_t)bh * SEQ * HEAD_DIM;
  const _Float16* Vt = Vth + (size_t)bh * HEAD_DIM * SEQ;
  const int q0 = qb * 64 + wave * 16;         // this wave's query tile

  const v16h qa0 = load_a_f16(Q, HEAD_DIM, q0, 0, lane);
  const v16h qa1 = load_a_f16(Q, HEAD_DIM, q0, 32, lane);

  v8f zero = {};
  v8f o[4] = {zero, zero, zero, zero};
  float mrow[8], lrow[8];
#pragma unroll
  for (int r = 0; r < 8; ++r) { mrow[r] = -1e30f; lrow[r] = 0.0f; }

  _Float16* myp = Pl[wave];
  const int half = lane >> 4;
  const int nl   = lane & 15;

  // Cooperative-staging index mapping (fixed per thread)
  const int krow = tid >> 2, kseg = (tid & 3) * 16;   // K: 32 rows x 4 segs
  const int vrow = tid >> 1, vseg = (tid & 1) * 16;   // V: 64 rows x 2 segs

  for (int key0 = 0; key0 < SEQ; key0 += 32) {
    if (key0 + 32 < SEQ) {                   // hint next chunk toward caches
      __builtin_prefetch(K  + (size_t)(key0 + 32) * HEAD_DIM, 0, 1);
      __builtin_prefetch(Vt + (size_t)vrow * SEQ + key0 + 32, 0, 1);
    }

    __syncthreads();                          // prior iter's LDS reads done
    {   // stage K chunk: Kl[key][d]
      const _Float16* src = K + (size_t)(key0 + krow) * HEAD_DIM + kseg;
      _Float16* dst = Kl + krow * KL_PITCH + kseg;
#pragma unroll
      for (int i = 0; i < 16; ++i) dst[i] = src[i];
    }
    {   // stage V^T chunk: Vl[d][t_local]
      const _Float16* src = Vt + (size_t)vrow * SEQ + key0 + vseg;
      _Float16* dst = Vl + vrow * VL_PITCH + vseg;
#pragma unroll
      for (int i = 0; i < 16; ++i) dst[i] = src[i];
    }
    __syncthreads();                          // staged data visible

    // S = Q * K^T from LDS (two 16-col tiles, Dh=64 -> two K-steps each)
    v8f s0 = zero, s1 = zero;
    {
      v16h bk = load_bT_f16(Kl, KL_PITCH, 0,  0,  lane);
      s0 = wmma_f16(qa0, bk, s0);
      bk = load_bT_f16(Kl, KL_PITCH, 32, 0,  lane);
      s0 = wmma_f16(qa1, bk, s0);
      bk = load_bT_f16(Kl, KL_PITCH, 0,  16, lane);
      s1 = wmma_f16(qa0, bk, s1);
      bk = load_bT_f16(Kl, KL_PITCH, 32, 16, lane);
      s1 = wmma_f16(qa1, bk, s1);
    }

    // Online softmax (C-layout: lane holds col nl, rows r+8*half)
    float sv0[8], sv1[8], rmax[8];
#pragma unroll
    for (int r = 0; r < 8; ++r) {
      sv0[r] = s0[r] * 0.125f;                // 1/sqrt(64)
      sv1[r] = s1[r] * 0.125f;
      rmax[r] = fmaxf(sv0[r], sv1[r]);
    }
#pragma unroll
    for (int msk = 1; msk < 16; msk <<= 1)
#pragma unroll
      for (int r = 0; r < 8; ++r)
        rmax[r] = fmaxf(rmax[r], __shfl_xor(rmax[r], msk, 32));

    float p0[8], p1[8], rsum[8];
#pragma unroll
    for (int r = 0; r < 8; ++r) {
      const float mnew = fmaxf(mrow[r], rmax[r]);
      const float corr = __expf(mrow[r] - mnew);
      mrow[r] = mnew;
      p0[r] = __expf(sv0[r] - mnew);
      p1[r] = __expf(sv1[r] - mnew);
      rsum[r] = p0[r] + p1[r];
      lrow[r] = lrow[r] * corr;
#pragma unroll
      for (int g = 0; g < 4; ++g) o[g][r] *= corr;
    }
#pragma unroll
    for (int msk = 1; msk < 16; msk <<= 1)
#pragma unroll
      for (int r = 0; r < 8; ++r)
        rsum[r] += __shfl_xor(rsum[r], msk, 32);
#pragma unroll
    for (int r = 0; r < 8; ++r) lrow[r] += rsum[r];

    // Re-layout P: C-fragment -> per-wave LDS (16x32) -> A-fragment
#pragma unroll
    for (int r = 0; r < 8; ++r) {
      const int row = r + 8 * half;
      myp[row * 32 + nl]      = (_Float16)p0[r];
      myp[row * 32 + 16 + nl] = (_Float16)p1[r];
    }
    __syncthreads();                          // (also fences wave-local P RAW)
    const v16h pa = load_a_f16(myp, 32, 0, 0, lane);

    // O += P * V from LDS (V stored transposed: contiguous-K B fragments)
#pragma unroll
    for (int g = 0; g < 4; ++g) {
      v16h bv = load_bT_f16(Vl, VL_PITCH, 0, g * 16, lane);
      o[g] = wmma_f16(pa, bv, o[g]);
    }
  }

  // Normalize and store f16 [B,T,C] with C index = h*64 + d
#pragma unroll
  for (int g = 0; g < 4; ++g) {
#pragma unroll
    for (int r = 0; r < 8; ++r) {
      const int t   = q0 + r + 8 * half;
      const int col = h * HEAD_DIM + g * 16 + nl;
      atth[((size_t)b * SEQ + t) * D_MODEL + col] = (_Float16)(o[g][r] / lrow[r]);
    }
  }
}

// ---------------------------------------------------------------------------
// Stage 4: output projection. One wave -> 16x64 tile, K = 1024, fp32 out.
// ---------------------------------------------------------------------------
__global__ void __launch_bounds__(256)
gemm_out_kernel(const _Float16* __restrict__ atth, const _Float16* __restrict__ wT,
                const float* __restrict__ bias, float* __restrict__ out) {
  const int lane = threadIdx.x & 31;
  const int wave = threadIdx.x >> 5;
  const int tile = blockIdx.x * 8 + wave;     // 8192 wave-tiles
  const int NT = D_MODEL / 64;                // 16
  const int mt = tile / NT, nt = tile - mt * NT;
  const int m0 = mt * 16, n0 = nt * 64;

  v8f zero = {};
  v8f acc[4] = {zero, zero, zero, zero};
  for (int k0 = 0; k0 < D_MODEL; k0 += 32) {
    v16h a = load_a_f16(atth, D_MODEL, m0, k0, lane);
#pragma unroll
    for (int g = 0; g < 4; ++g) {
      v16h b = load_bT_f16(wT, D_MODEL, k0, n0 + g * 16, lane);
      acc[g] = wmma_f16(a, b, acc[g]);
    }
  }

  const int half = lane >> 4;
  const int nl   = lane & 15;
#pragma unroll
  for (int g = 0; g < 4; ++g) {
    const int n = n0 + g * 16 + nl;
    const float bv = bias[n];
#pragma unroll
    for (int r = 0; r < 8; ++r) {
      const int m = m0 + r + 8 * half;
      out[(size_t)m * D_MODEL + n] = acc[g][r] + bv;
    }
  }
}

// ---------------------------------------------------------------------------
extern "C" void kernel_launch(void* const* d_in, const int* in_sizes, int n_in,
                              void* d_out, int out_size, void* d_ws, size_t ws_size,
                              hipStream_t stream) {
  const float* x     = (const float*)d_in[0];   // [8192,1024]
  const float* w_qkv = (const float*)d_in[1];   // [1024,3072]
  const float* b_qkv = (const float*)d_in[2];   // [3072]
  const float* w_out = (const float*)d_in[3];   // [1024,1024]
  const float* b_out = (const float*)d_in[4];   // [1024]
  float* out = (float*)d_out;                   // [8192,1024]

  char* ws = (char*)d_ws;
  size_t off = 0;
  auto carve = [&](size_t elems) {
    _Float16* p = (_Float16*)(ws + off);
    off = (off + elems * sizeof(_Float16) + 255) & ~(size_t)255;
    return p;
  };
  _Float16* wqkvT = carve((size_t)N_QKV * D_MODEL);           // [3072,1024]
  _Float16* woutT = carve((size_t)D_MODEL * D_MODEL);         // [1024,1024]
  _Float16* Qh    = carve((size_t)BATCH * NUM_HEADS * SEQ * HEAD_DIM);
  _Float16* Kh    = carve((size_t)BATCH * NUM_HEADS * SEQ * HEAD_DIM);
  _Float16* Vth   = carve((size_t)BATCH * NUM_HEADS * HEAD_DIM * SEQ);
  _Float16* atth  = carve((size_t)ROWS * D_MODEL);            // [8192,1024]
  (void)ws_size; (void)in_sizes; (void)n_in; (void)out_size;

  // Stage 1: weight casts/transposes
  {
    int n1 = D_MODEL * N_QKV;
    transpose_f32_to_f16<<<(n1 + 255) / 256, 256, 0, stream>>>(w_qkv, wqkvT,
                                                               D_MODEL, N_QKV);
    int n2 = D_MODEL * D_MODEL;
    transpose_f32_to_f16<<<(n2 + 255) / 256, 256, 0, stream>>>(w_out, woutT,
                                                               D_MODEL, D_MODEL);
  }
  // Stage 2: QKV projection (24576 wave-tiles / 8 waves per block)
  gemm_qkv_kernel<<<3072, 256, 0, stream>>>(x, wqkvT, b_qkv, Qh, Kh, Vth);
  // Stage 3: flash attention (64 bh x 32 query-blocks, 4 waves per block)
  attn_kernel<<<2048, 128, 0, stream>>>(Qh, Kh, Vth, atth);
  // Stage 4: output projection (8192 wave-tiles / 8 waves per block)
  gemm_out_kernel<<<1024, 256, 0, stream>>>(atth, woutT, b_out, out);
}
